// LlamaDifferentialAttention_60017872995014
// MI455X (gfx1250) — matrix operand order repaired and checked by reading, not compiled, not run on hardware
//
#include <hip/hip_runtime.h>
#include <math.h>

#define NH 32
#define NKV 8
#define HD 64
#define BATCH 2
#define SEQ 1024
#define HID 2048
// 0.8 - 0.6*exp(-0.3*12)
#define LAMBDA_INIT 0.7836057665316244f

typedef __attribute__((ext_vector_type(16))) __bf16 v16bf;
typedef __attribute__((ext_vector_type(8)))  __bf16 v8bf;
typedef __attribute__((ext_vector_type(4)))  __bf16 v4bf;
typedef __attribute__((ext_vector_type(8)))  float  v8f;
typedef __attribute__((ext_vector_type(4)))  float  v4f;

// Assemble a 16-element bf16 WMMA fragment from two 8-element (16B) runs.
static __device__ __forceinline__ v16bf frag_from(const __bf16* p0, const __bf16* p1) {
    v8bf lo = *(const v8bf*)p0;
    v8bf hi = *(const v8bf*)p1;
    v16bf r;
#pragma unroll
    for (int i = 0; i < 8; ++i) { r[i] = lo[i]; r[i + 8] = hi[i]; }
    return r;
}

// ---------------------------------------------------------------------------
// Generic C[M,N] = X[M,K] * W[N,K]^T  (f32 in/out, bf16 WMMA compute)
// block = 128 threads (4 waves), tile 64x64.
// Double-buffered LDS: one barrier per K-chunk; the next chunk's global
// loads are issued before the barrier so their latency hides under the
// WMMA burst of the current chunk.
// ---------------------------------------------------------------------------
__global__ __launch_bounds__(128)
void gemm_xwT_wmma(const float* __restrict__ X, const float* __restrict__ W,
                   float* __restrict__ C, int M, int N, int K) {
    __shared__ __bf16 xs[2][64][64];
    __shared__ __bf16 wsm[2][64][64];
    const int tid  = threadIdx.x;
    const int wave = tid >> 5;
    const int lane = tid & 31;
    const int hi   = lane >> 4;
    const int l16  = lane & 15;
    const int n0 = blockIdx.x * 64;
    const int m0 = blockIdx.y * 64;

    v8f acc[4] = {};
    v4f xr[8], wr[8];

    // fetch + store chunk 0 into buffer 0
#pragma unroll
    for (int i = 0; i < 8; ++i) {
        const int vid = tid + i * 128;
        const int r = vid >> 4, c4 = (vid & 15) << 2;
        xr[i] = *(const v4f*)&X[(size_t)(m0 + r) * K + c4];
        wr[i] = *(const v4f*)&W[(size_t)(n0 + r) * K + c4];
    }
#pragma unroll
    for (int i = 0; i < 8; ++i) {
        const int vid = tid + i * 128;
        const int r = vid >> 4, c4 = (vid & 15) << 2;
        v4bf xb, wb;
#pragma unroll
        for (int j = 0; j < 4; ++j) { xb[j] = (__bf16)xr[i][j]; wb[j] = (__bf16)wr[i][j]; }
        *(v4bf*)&xs[0][r][c4]  = xb;
        *(v4bf*)&wsm[0][r][c4] = wb;
    }

    const int nk = K >> 6;
    for (int kc = 0; kc < nk; ++kc) {
        const int cur = kc & 1, nxt = cur ^ 1;
        if (kc + 1 < nk) {  // issue next chunk's global loads now
            const int kb = (kc + 1) << 6;
#pragma unroll
            for (int i = 0; i < 8; ++i) {
                const int vid = tid + i * 128;
                const int r = vid >> 4, c4 = (vid & 15) << 2;
                xr[i] = *(const v4f*)&X[(size_t)(m0 + r) * K + kb + c4];
                wr[i] = *(const v4f*)&W[(size_t)(n0 + r) * K + kb + c4];
            }
        }
        __syncthreads();  // buf[cur] fully stored; buf[nxt] no longer read
#pragma unroll
        for (int c = 0; c < 2; ++c) {
            const int aoff = c * 32 + (hi ? 16 : 0);    // A: contiguous 16-run
            const __bf16* ar = &xs[cur][wave * 16 + l16][0];
            const v16bf a = frag_from(ar + aoff, ar + aoff + 8);
            const int boff = c * 32 + (hi ? 16 : 0);    // B: contiguous 16-run
            v16bf bfr[4];
#pragma unroll
            for (int t = 0; t < 4; ++t) {
                const __bf16* br = &wsm[cur][t * 16 + l16][0];
                bfr[t] = frag_from(br + boff, br + boff + 8);
            }
#pragma unroll
            for (int t = 0; t < 4; ++t)
                acc[t] = __builtin_amdgcn_wmma_f32_16x16x32_bf16(
                    false, a, false, bfr[t], (short)0, acc[t], false, false);
        }
        if (kc + 1 < nk) {  // convert + store next chunk (nobody reads buf[nxt] yet)
#pragma unroll
            for (int i = 0; i < 8; ++i) {
                const int vid = tid + i * 128;
                const int r = vid >> 4, c4 = (vid & 15) << 2;
                v4bf xb, wb;
#pragma unroll
                for (int j = 0; j < 4; ++j) { xb[j] = (__bf16)xr[i][j]; wb[j] = (__bf16)wr[i][j]; }
                *(v4bf*)&xs[nxt][r][c4]  = xb;
                *(v4bf*)&wsm[nxt][r][c4] = wb;
            }
        }
    }
#pragma unroll
    for (int t = 0; t < 4; ++t)
#pragma unroll
        for (int j = 0; j < 8; ++j) {
            const int row = m0 + wave * 16 + j + hi * 8;
            const int col = n0 + t * 16 + l16;
            C[(size_t)row * N + col] = acc[t][j];
        }
}

// ---------------------------------------------------------------------------
// RoPE + pack f32 -> bf16 [b][h][s][d]
// ---------------------------------------------------------------------------
__global__ void rope_pack_q(const float* __restrict__ qp,
                            __bf16* __restrict__ q1b, __bf16* __restrict__ q2b) {
    const int t = blockIdx.x * blockDim.x + threadIdx.x; // B*S*NH*32
    const int d = t & 31;
    const int h = (t >> 5) & 31;
    const int s = (t >> 10) & (SEQ - 1);
    const int b = t >> 20;
    const size_t row = (size_t)(b * SEQ + s) * (2 * HID);
    const float ang = (float)s * __expf(-(2.0f * d / 64.0f) * 9.210340371976184f);
    const float cs = cosf(ang), sn = sinf(ang);
    const size_t o = ((size_t)(b * NH + h) * SEQ + s) * HD;

    float x0 = qp[row + h * HD + d];
    float x1 = qp[row + h * HD + d + 32];
    q1b[o + d]      = (__bf16)(x0 * cs - x1 * sn);
    q1b[o + d + 32] = (__bf16)(x1 * cs + x0 * sn);
    x0 = qp[row + HID + h * HD + d];
    x1 = qp[row + HID + h * HD + d + 32];
    q2b[o + d]      = (__bf16)(x0 * cs - x1 * sn);
    q2b[o + d + 32] = (__bf16)(x1 * cs + x0 * sn);
}

__global__ void rope_pack_k(const float* __restrict__ kp,
                            __bf16* __restrict__ k1b, __bf16* __restrict__ k2b) {
    const int t = blockIdx.x * blockDim.x + threadIdx.x; // B*S*NKV*32
    const int d = t & 31;
    const int h = (t >> 5) & 7;
    const int s = (t >> 8) & (SEQ - 1);
    const int b = t >> 18;
    const size_t row = (size_t)(b * SEQ + s) * (2 * NKV * HD);
    const float ang = (float)s * __expf(-(2.0f * d / 64.0f) * 9.210340371976184f);
    const float cs = cosf(ang), sn = sinf(ang);
    const size_t o = ((size_t)(b * NKV + h) * SEQ + s) * HD;

    float x0 = kp[row + h * HD + d];
    float x1 = kp[row + h * HD + d + 32];
    k1b[o + d]      = (__bf16)(x0 * cs - x1 * sn);
    k1b[o + d + 32] = (__bf16)(x1 * cs + x0 * sn);
    x0 = kp[row + NKV * HD + h * HD + d];
    x1 = kp[row + NKV * HD + h * HD + d + 32];
    k2b[o + d]      = (__bf16)(x0 * cs - x1 * sn);
    k2b[o + d + 32] = (__bf16)(x1 * cs + x0 * sn);
}

__global__ void conv_v(const float* __restrict__ vp, __bf16* __restrict__ vb) {
    const int t = blockIdx.x * blockDim.x + threadIdx.x; // B*S*NKV*64
    const int d = t & 63;
    const int h = (t >> 6) & 7;
    const int s = (t >> 9) & (SEQ - 1);
    const int b = t >> 19;
    vb[((size_t)(b * NKV + h) * SEQ + s) * HD + d] =
        (__bf16)vp[(size_t)(b * SEQ + s) * (NKV * HD) + h * HD + d];
}

// ---------------------------------------------------------------------------
// Differential flash attention. block = 128 (4 waves); each wave owns a
// 16-row q tile; K1/K2/V(T) staged in LDS per 64-key tile; two independent
// online softmaxes; all matmuls on v_wmma_f32_16x16x32_bf16.
// ---------------------------------------------------------------------------
__global__ __launch_bounds__(128)
void diff_attn_wmma(const __bf16* __restrict__ q1b, const __bf16* __restrict__ q2b,
                    const __bf16* __restrict__ k1b, const __bf16* __restrict__ k2b,
                    const __bf16* __restrict__ vb,
                    const float* __restrict__ lq1, const float* __restrict__ lk1,
                    const float* __restrict__ lq2, const float* __restrict__ lk2,
                    float* __restrict__ attn) {
    __shared__ __bf16 ks[2][64][64];   // [stream][key][dim]
    __shared__ __bf16 vts[64][64];     // transposed: [dim][key]
    __shared__ __bf16 ps[4][16][64];   // per-wave probability tile

    const int tid  = threadIdx.x;
    const int wave = tid >> 5;
    const int lane = tid & 31;
    const int hi   = lane >> 4;
    const int l16  = lane & 15;
    const int q0   = blockIdx.x * 64;
    const int h    = blockIdx.y;
    const int b    = blockIdx.z;
    const int kvh  = h >> 2;

    // lambda_full = exp(sum lq1*lk1) - exp(sum lq2*lk2) + LAMBDA_INIT
    float sl1 = lq1[lane] * lk1[lane] + lq1[lane + 32] * lk1[lane + 32];
    float sl2 = lq2[lane] * lk2[lane] + lq2[lane + 32] * lk2[lane + 32];
#pragma unroll
    for (int s = 1; s < 32; s <<= 1) {
        sl1 += __shfl_xor(sl1, s, 32);
        sl2 += __shfl_xor(sl2, s, 32);
    }
    const float lam_full = expf(sl1) - expf(sl2) + LAMBDA_INIT;

    // Q fragments (A matrices), both streams, both 32-wide K chunks
    v16bf qf[2][2];
    {
        const size_t qrow = ((size_t)(b * NH + h) * SEQ + q0 + wave * 16 + l16) * HD;
        const __bf16* q1p = q1b + qrow;
        const __bf16* q2p = q2b + qrow;
#pragma unroll
        for (int c = 0; c < 2; ++c) {
            const int off = c * 32 + (hi ? 16 : 0);
            qf[0][c] = frag_from(q1p + off, q1p + off + 8);
            qf[1][c] = frag_from(q2p + off, q2p + off + 8);
        }
    }

    v8f oacc[2][4] = {};
    float mrow[2][8], lrow[2][8];
#pragma unroll
    for (int st = 0; st < 2; ++st)
#pragma unroll
        for (int j = 0; j < 8; ++j) { mrow[st][j] = -1e30f; lrow[st][j] = 0.0f; }

    const int nkt = (q0 >> 6) + 1;     // causal: key tiles 0 .. q0/64
    const float scale = 0.125f;        // 1/sqrt(64)

    for (int kt = 0; kt < nkt; ++kt) {
        __syncthreads();
        const size_t kbase = ((size_t)(b * NKV + kvh) * SEQ + kt * 64) * HD;
#pragma unroll
        for (int i = 0; i < 8; ++i) {
            const int vid = tid + i * 128;
            const int key = vid >> 4;
            const int c4  = (vid & 15) << 2;
            *(v4bf*)&ks[0][key][c4] = *(const v4bf*)&k1b[kbase + key * HD + c4];
            *(v4bf*)&ks[1][key][c4] = *(const v4bf*)&k2b[kbase + key * HD + c4];
            v4bf v4 = *(const v4bf*)&vb[kbase + key * HD + c4];
#pragma unroll
            for (int j = 0; j < 4; ++j) vts[c4 + j][key] = v4[j];
        }
        __syncthreads();

        for (int st = 0; st < 2; ++st) {
            // --- scores for the full 64-key tile, kept in registers ---
            v8f  sc[4];
            float tm[8];
#pragma unroll
            for (int j = 0; j < 8; ++j) tm[j] = -1e30f;
#pragma unroll
            for (int sub = 0; sub < 4; ++sub) {
                v8f c = {};
#pragma unroll
                for (int cc = 0; cc < 2; ++cc) {
                    const int boff = cc * 32 + (hi ? 16 : 0);
                    const __bf16* kr = &ks[st][sub * 16 + l16][0];
                    v16bf bk = frag_from(kr + boff, kr + boff + 8);
                    c = __builtin_amdgcn_wmma_f32_16x16x32_bf16(
                        false, qf[st][cc], false, bk, (short)0, c, false, false);
                }
                const int kidx = kt * 64 + sub * 16 + l16;
#pragma unroll
                for (int j = 0; j < 8; ++j) {
                    const int qidx = q0 + wave * 16 + j + hi * 8;
                    const float v = (kidx <= qidx) ? c[j] * scale : -1e30f;
                    c[j] = v;
                    tm[j] = fmaxf(tm[j], v);
                }
                sc[sub] = c;
            }
            // row max across the 16 key lanes
#pragma unroll
            for (int s = 1; s < 16; s <<= 1)
#pragma unroll
                for (int j = 0; j < 8; ++j)
                    tm[j] = fmaxf(tm[j], __shfl_xor(tm[j], s, 32));

            float alpha[8], rs[8];
#pragma unroll
            for (int j = 0; j < 8; ++j) {
                const float nm = fmaxf(mrow[st][j], tm[j]);
                alpha[j] = __expf(mrow[st][j] - nm);
                mrow[st][j] = nm;
                rs[j] = 0.0f;
            }
#pragma unroll
            for (int sub = 0; sub < 4; ++sub)
#pragma unroll
                for (int j = 0; j < 8; ++j) {
                    const float p = __expf(sc[sub][j] - mrow[st][j]);
                    rs[j] += p;
                    ps[wave][j + hi * 8][sub * 16 + l16] = (__bf16)p;
                }
#pragma unroll
            for (int s = 1; s < 16; s <<= 1)
#pragma unroll
                for (int j = 0; j < 8; ++j)
                    rs[j] += __shfl_xor(rs[j], s, 32);
#pragma unroll
            for (int j = 0; j < 8; ++j)
                lrow[st][j] = lrow[st][j] * alpha[j] + rs[j];
#pragma unroll
            for (int t = 0; t < 4; ++t)
#pragma unroll
                for (int j = 0; j < 8; ++j)
                    oacc[st][t][j] *= alpha[j];

            // make per-wave prob writes visible to our own ds reads
            asm volatile("s_wait_dscnt 0" ::: "memory");

            // --- P @ V ---
#pragma unroll
            for (int cc = 0; cc < 2; ++cc) {
                const int aoff = cc * 32 + (hi ? 16 : 0);
                const __bf16* pr = &ps[wave][l16][0];
                v16bf pa = frag_from(pr + aoff, pr + aoff + 8);
                const int boff = cc * 32 + (hi ? 16 : 0);
                v16bf bvr[4];
#pragma unroll
                for (int t = 0; t < 4; ++t) {
                    const __bf16* vr = &vts[t * 16 + l16][0];
                    bvr[t] = frag_from(vr + boff, vr + boff + 8);
                }
#pragma unroll
                for (int t = 0; t < 4; ++t)
                    oacc[st][t] = __builtin_amdgcn_wmma_f32_16x16x32_bf16(
                        false, pa, false, bvr[t], (short)0, oacc[st][t], false, false);
            }
        }
    }

    const float om = 1.0f - LAMBDA_INIT;
#pragma unroll
    for (int t = 0; t < 4; ++t)
#pragma unroll
        for (int j = 0; j < 8; ++j) {
            const int q = q0 + wave * 16 + j + hi * 8;
            const float v = (oacc[0][t][j] / lrow[0][j]
                             - lam_full * oacc[1][t][j] / lrow[1][j]) * om;
            attn[(size_t)(b * SEQ + q) * HID + h * HD + t * 16 + l16] = v;
        }
}

// ---------------------------------------------------------------------------
extern "C" void kernel_launch(void* const* d_in, const int* in_sizes, int n_in,
                              void* d_out, int out_size, void* d_ws, size_t ws_size,
                              hipStream_t stream) {
    const float* hs  = (const float*)d_in[0];
    const float* wq  = (const float*)d_in[1];
    const float* wk  = (const float*)d_in[2];
    const float* wv  = (const float*)d_in[3];
    const float* wo  = (const float*)d_in[4];
    const float* lq1 = (const float*)d_in[5];
    const float* lk1 = (const float*)d_in[6];
    const float* lq2 = (const float*)d_in[7];
    const float* lk2 = (const float*)d_in[8];

    char* ws = (char*)d_ws;
    size_t off = 0;
    float*  qp   = (float*)(ws + off);  off += (size_t)2048 * 4096 * 4;
    float*  kp   = (float*)(ws + off);  off += (size_t)2048 * 1024 * 4;
    float*  vp   = (float*)(ws + off);  off += (size_t)2048 * 512 * 4;
    __bf16* q1b  = (__bf16*)(ws + off); off += (size_t)2048 * 2048 * 2;
    __bf16* q2b  = (__bf16*)(ws + off); off += (size_t)2048 * 2048 * 2;
    __bf16* k1b  = (__bf16*)(ws + off); off += (size_t)2048 * 512 * 2;
    __bf16* k2b  = (__bf16*)(ws + off); off += (size_t)2048 * 512 * 2;
    __bf16* vbuf = (__bf16*)(ws + off); off += (size_t)2048 * 512 * 2;
    float*  attn = (float*)(ws + off);  off += (size_t)2048 * 2048 * 4;

    // projections
    gemm_xwT_wmma<<<dim3(64, 32), 128, 0, stream>>>(hs, wq, qp, 2048, 4096, 2048);
    gemm_xwT_wmma<<<dim3(16, 32), 128, 0, stream>>>(hs, wk, kp, 2048, 1024, 2048);
    gemm_xwT_wmma<<<dim3(8, 32),  128, 0, stream>>>(hs, wv, vp, 2048, 512, 2048);

    // rope + bf16 packing
    rope_pack_q<<<8192, 256, 0, stream>>>(qp, q1b, q2b);
    rope_pack_k<<<2048, 256, 0, stream>>>(kp, k1b, k2b);
    conv_v<<<4096, 256, 0, stream>>>(vp, vbuf);

    // differential flash attention
    diff_attn_wmma<<<dim3(16, 32, 2), 128, 0, stream>>>(
        q1b, q2b, k1b, k2b, vbuf, lq1, lk1, lq2, lk2, attn);

    // output projection
    gemm_xwT_wmma<<<dim3(32, 32), 128, 0, stream>>>(attn, wo, (float*)d_out,
                                                    2048, 2048, 2048);
}
